// GCN_20375324852677
// MI455X (gfx1250) — matrix operand order repaired and checked by reading, not compiled, not run on hardware
//
#include <hip/hip_runtime.h>
#include <hip/hip_bf16.h>

typedef __attribute__((ext_vector_type(16))) __bf16 v16bf;
typedef __attribute__((ext_vector_type(8)))  float  v8f;

// round-to-nearest-even fp32 -> bf16 (used only in one-shot weight swizzle)
__device__ __forceinline__ __bf16 f2bf(float f) {
    unsigned u = __float_as_uint(f);
    u += 0x7FFFu + ((u >> 16) & 1u);
    unsigned short h = (unsigned short)(u >> 16);
    __bf16 r;
    __builtin_memcpy(&r, &h, sizeof(r));
    return r;
}

// pack two fp32 into two truncated bf16 in one v_perm_b32 (lo -> bits[15:0])
__device__ __forceinline__ unsigned pkbf(float lo, float hi) {
    return __builtin_amdgcn_perm(__float_as_uint(hi), __float_as_uint(lo), 0x07060302u);
}

// ---------------- degree / norm ----------------
__global__ void k_init_deg(float* deg, int n) {
    int i = blockIdx.x * blockDim.x + threadIdx.x;
    if (i < n) deg[i] = 1.0f;                  // self-loop weight
}

__global__ void k_deg_accum(const long long* __restrict__ dst,
                            const float* __restrict__ ew,
                            float* __restrict__ deg, long long E) {
    long long e = (long long)blockIdx.x * blockDim.x + threadIdx.x;
    if (e < E) atomicAdd(&deg[(int)dst[e]], ew[e]);
}

__global__ void k_dinv(float* deg, int n) {
    int i = blockIdx.x * blockDim.x + threadIdx.x;
    if (i < n) {
        float d = deg[i];
        deg[i] = (d > 0.0f) ? rsqrtf(d) : 0.0f;   // in-place: deg -> dinv
    }
}

// ---------------- weight pre-swizzle into WMMA B-operand layout ----------------
// out[((kt*NT + tn)*32 + lane)*16 + e] = bf16( W[(kt*32 + 8*(lane>>4) + (e&7) + 16*(e>>3))*Nc + tn*16 + (lane&15)] )
// zero-padded for col >= Nc.  One thread per (kt, tn, lane); 16 bf16 (32B) per thread.
__global__ void k_bswizzle(const float* __restrict__ W, __bf16* __restrict__ out,
                           int K, int Nc, int NT) {
    int t = blockIdx.x * blockDim.x + threadIdx.x;
    int total = (K >> 5) * NT * 32;
    if (t >= total) return;
    int lane = t & 31;
    int tn   = (t >> 5) % NT;
    int kt   = t / (32 * NT);
    int half = lane >> 4;
    int l16  = lane & 15;
    int col  = tn * 16 + l16;
    __bf16* o = out + (size_t)t * 16;
    #pragma unroll
    for (int e = 0; e < 16; ++e) {
        int k = kt * 32 + 8 * half + (e & 7) + 16 * (e >> 3);
        float v = (col < Nc) ? W[(size_t)k * Nc + col] : 0.0f;
        o[e] = f2bf(v);
    }
}

// ---------------- WMMA GEMM: C[M,NC] = A[M,K] * Bsw ----------------
// one wave per 16-row stripe, NT accumulators covering all NT*16 columns.
// A converted fp32->bf16 via v_perm_b32 pairs; B loaded pre-swizzled (32B vector per tile).
template <int NT, int NC>
__global__ void k_gemm_wmma(const float* __restrict__ A, const __bf16* __restrict__ Bsw,
                            float* __restrict__ C, int M, int K) {
    const int wave = threadIdx.x >> 5;
    const int lane = threadIdx.x & 31;
    const int tm   = blockIdx.x * (blockDim.x >> 5) + wave;
    if (tm * 16 >= M) return;                  // wave-uniform: EXEC stays all-1s
    const int half = lane >> 4;                // 0: lanes 0-15, 1: lanes 16-31
    const int l16  = lane & 15;

    int row  = tm * 16 + l16;
    int arow = (row < M) ? row : (M - 1);      // clamp for partial last tile
    const float* arp = A + (size_t)arow * K + 8 * half;

    v8f acc[NT];
    #pragma unroll
    for (int t = 0; t < NT; ++t) acc[t] = v8f{};

    const int nkt = K >> 5;
    for (int kt = 0; kt < nkt; ++kt) {
        // ---- A operand (16x32 bf16): e0-7 -> K=kk+8*half+e ; e8-15 -> K=kk+16+8*half+(e&7)
        const float* ap = arp + kt * 32;
        float4 a0 = *(const float4*)(ap + 0);
        float4 a1 = *(const float4*)(ap + 4);
        float4 a2 = *(const float4*)(ap + 16);
        float4 a3 = *(const float4*)(ap + 20);
        union { unsigned u[8]; v16bf v; } av;
        av.u[0] = pkbf(a0.x, a0.y);
        av.u[1] = pkbf(a0.z, a0.w);
        av.u[2] = pkbf(a1.x, a1.y);
        av.u[3] = pkbf(a1.z, a1.w);
        av.u[4] = pkbf(a2.x, a2.y);
        av.u[5] = pkbf(a2.z, a2.w);
        av.u[6] = pkbf(a3.x, a3.y);
        av.u[7] = pkbf(a3.z, a3.w);

        const __bf16* bp = Bsw + ((size_t)kt * NT * 32 + lane) * 16;
        #pragma unroll
        for (int tn = 0; tn < NT; ++tn) {
            v16bf bv = *(const v16bf*)(bp + (size_t)tn * 32 * 16);
            acc[tn] = __builtin_amdgcn_wmma_f32_16x16x32_bf16(
                          false, av.v, false, bv, (short)0, acc[tn], false, false);
        }
    }

    // ---- store: D layout VGPR v -> row tm*16 + v + 8*half, col tn*16 + l16
    float* cbase = C + (size_t)(tm * 16 + 8 * half) * NC;
    if ((tm + 1) * 16 <= M) {                  // full stripe: unguarded rows
        #pragma unroll
        for (int tn = 0; tn < NT; ++tn) {
            int col = tn * 16 + l16;
            if (col < NC) {                    // constexpr-eliminated except last tile
                #pragma unroll
                for (int v = 0; v < 8; ++v)
                    cbase[(size_t)v * NC + col] = acc[tn][v];
            }
        }
    } else {
        #pragma unroll
        for (int tn = 0; tn < NT; ++tn) {
            int col = tn * 16 + l16;
            if (col < NC) {
                #pragma unroll
                for (int v = 0; v < 8; ++v) {
                    int r = tm * 16 + v + 8 * half;
                    if (r < M) C[(size_t)r * NC + col] = acc[tn][v];
                }
            }
        }
    }
}

// ---------------- aggregation ----------------
// out[n,f] = dinv[n]^2 * h[n,f] + bias[f]   (self-loop term, norm = dinv*1*dinv)
__global__ void k_agg_init(const float* __restrict__ h, const float* __restrict__ dinv,
                           const float* __restrict__ bias, float* __restrict__ out,
                           long long total, int F) {
    long long i = (long long)blockIdx.x * blockDim.x + threadIdx.x;
    if (i < total) {
        int node = (int)(i / F);
        int f    = (int)(i % F);
        float d  = dinv[node];
        out[i] = d * d * h[i] + bias[f];
    }
}

// F=64 scatter: 16 threads per edge, 4 floats each
__global__ void k_agg_edge64(const long long* __restrict__ src, const long long* __restrict__ dst,
                             const float* __restrict__ ew, const float* __restrict__ dinv,
                             const float* __restrict__ h, float* __restrict__ out, long long E) {
    long long t = (long long)blockIdx.x * blockDim.x + threadIdx.x;
    long long e = t >> 4;
    int f4 = (int)(t & 15) * 4;
    if (e < E) {
        int s = (int)src[e], d = (int)dst[e];
        float nrm = dinv[s] * ew[e] * dinv[d];
        float4 v = *(const float4*)(h + (size_t)s * 64 + f4);
        float* o = out + (size_t)d * 64 + f4;
        atomicAdd(o + 0, nrm * v.x);
        atomicAdd(o + 1, nrm * v.y);
        atomicAdd(o + 2, nrm * v.z);
        atomicAdd(o + 3, nrm * v.w);
    }
}

// F=40 scatter: one thread per (edge, class)
__global__ void k_agg_edge40(const long long* __restrict__ src, const long long* __restrict__ dst,
                             const float* __restrict__ ew, const float* __restrict__ dinv,
                             const float* __restrict__ h, float* __restrict__ out, long long E) {
    long long t = (long long)blockIdx.x * blockDim.x + threadIdx.x;
    long long e = t / 40;
    int c = (int)(t % 40);
    if (e < E) {
        int s = (int)src[e], d = (int)dst[e];
        float nrm = dinv[s] * ew[e] * dinv[d];
        atomicAdd(&out[(size_t)d * 40 + c], nrm * h[(size_t)s * 40 + c]);
    }
}

__global__ void k_relu(float* x, long long n) {
    long long i = (long long)blockIdx.x * blockDim.x + threadIdx.x;
    if (i < n) x[i] = fmaxf(x[i], 0.0f);
}

// in-place row-wise log_softmax over 40 classes
__global__ void k_logsoftmax40(float* __restrict__ x, int n) {
    int i = blockIdx.x * blockDim.x + threadIdx.x;
    if (i < n) {
        float* r = x + (size_t)i * 40;
        float m = r[0];
        #pragma unroll
        for (int c = 1; c < 40; ++c) m = fmaxf(m, r[c]);
        float s = 0.0f;
        #pragma unroll
        for (int c = 0; c < 40; ++c) s += __expf(r[c] - m);
        float lse = m + __logf(s);
        #pragma unroll
        for (int c = 0; c < 40; ++c) r[c] -= lse;
    }
}

extern "C" void kernel_launch(void* const* d_in, const int* in_sizes, int n_in,
                              void* d_out, int out_size, void* d_ws, size_t ws_size,
                              hipStream_t stream) {
    const float*     x  = (const float*)d_in[0];
    const long long* ei = (const long long*)d_in[1];   // int64 [2, E]
    const float*     ew = (const float*)d_in[2];
    const float*     W1 = (const float*)d_in[3];
    const float*     b1 = (const float*)d_in[4];
    const float*     W2 = (const float*)d_in[5];
    const float*     b2 = (const float*)d_in[6];
    float*           out = (float*)d_out;

    const int F = 512, H = 64, C = 40;
    const int N = in_sizes[0] / F;
    const long long E = in_sizes[2];
    const long long* src = ei;
    const long long* dst = ei + E;

    const int NT1 = 4;                 // 64 cols  -> 4 x 16 tiles
    const int NT2 = 3;                 // 40 cols  -> 3 x 16 tiles (guarded)

    // workspace carve-out
    char* ws = (char*)d_ws;
    size_t off = 0;
    auto carve = [&](size_t bytes) {
        void* p = ws + off;
        off += (bytes + 255) & ~(size_t)255;
        return p;
    };
    float*  dinv = (float*)carve((size_t)N * sizeof(float));      // deg -> dinv
    float*  h1   = (float*)carve((size_t)N * H * sizeof(float));
    float*  h1a  = (float*)carve((size_t)N * H * sizeof(float));
    float*  h2   = (float*)carve((size_t)N * C * sizeof(float));
    __bf16* W1sw = (__bf16*)carve((size_t)(F / 32) * NT1 * 32 * 16 * sizeof(__bf16));
    __bf16* W2sw = (__bf16*)carve((size_t)(H / 32) * NT2 * 32 * 16 * sizeof(__bf16));

    const int TB = 256;
    auto nblk = [](long long work, int tb) { return (int)((work + tb - 1) / tb); };

    // 1) degree + norm
    k_init_deg <<<nblk(N, TB), TB, 0, stream>>>(dinv, N);
    k_deg_accum<<<nblk(E, TB), TB, 0, stream>>>(dst, ew, dinv, E);
    k_dinv     <<<nblk(N, TB), TB, 0, stream>>>(dinv, N);

    // 2) pre-swizzle weights into WMMA B-operand layout (bf16)
    k_bswizzle<<<nblk((long long)(F / 32) * NT1 * 32, TB), TB, 0, stream>>>(W1, W1sw, F, H, NT1);
    k_bswizzle<<<nblk((long long)(H / 32) * NT2 * 32, TB), TB, 0, stream>>>(W2, W2sw, H, C, NT2);

    // 3) h1 = x @ W1   (WMMA, 16-row stripe per wave, 4 col tiles)
    {
        int tm = (N + 15) / 16;
        int blocks = (tm + 7) / 8;                     // 8 waves/block
        k_gemm_wmma<NT1, 64><<<blocks, TB, 0, stream>>>(x, W1sw, h1, N, F);
    }

    // 4) h1a = segment_sum(norm * h1[src]) + b1 ; relu
    k_agg_init  <<<nblk((long long)N * H, TB), TB, 0, stream>>>(h1, dinv, b1, h1a, (long long)N * H, H);
    k_agg_edge64<<<nblk(E * 16, TB), TB, 0, stream>>>(src, dst, ew, dinv, h1, h1a, E);
    k_relu      <<<nblk((long long)N * H, TB), TB, 0, stream>>>(h1a, (long long)N * H);

    // 5) h2 = h1a @ W2   (WMMA, 3 col tiles, last tile store-guarded)
    {
        int tm = (N + 15) / 16;
        int blocks = (tm + 7) / 8;
        k_gemm_wmma<NT2, 40><<<blocks, TB, 0, stream>>>(h1a, W2sw, h2, N, H);
    }

    // 6) out = segment_sum(norm * h2[src]) + b2 ; log_softmax (in-place on d_out)
    k_agg_init  <<<nblk((long long)N * C, TB), TB, 0, stream>>>(h2, dinv, b2, out, (long long)N * C, C);
    k_agg_edge40<<<nblk(E * 40, TB), TB, 0, stream>>>(src, dst, ew, dinv, h2, out, E);
    k_logsoftmax40<<<nblk(N, TB), TB, 0, stream>>>(out, N);
}
